// GCN_39908836114941
// MI455X (gfx1250) — matrix operand order repaired and checked by reading, not compiled, not run on hardware
//
#include <hip/hip_runtime.h>
#include <stdint.h>

// ---------------------------------------------------------------------------
// Types for CDNA5 WMMA (wave32, v_wmma_f32_16x16x32_bf16)
// ---------------------------------------------------------------------------
typedef __attribute__((ext_vector_type(16))) __bf16 v16bf;
typedef __attribute__((ext_vector_type(8)))  __bf16 v8bf;
typedef __attribute__((ext_vector_type(4)))  __bf16 v4bf;
typedef __attribute__((ext_vector_type(8)))  float  v8f;

__device__ __forceinline__ __bf16 f2bf(float f) {
  uint32_t u   = __builtin_bit_cast(uint32_t, f);
  uint32_t lsb = (u >> 16) & 1u;
  u += 0x7fffu + lsb;                     // round-to-nearest-even
  uint16_t h = (uint16_t)(u >> 16);
  return __builtin_bit_cast(__bf16, h);
}

// ---------------------------------------------------------------------------
// gcn_norm: deg (init 1.0 for self loop), += w at dst, rsqrt in place, norm[e]
// ---------------------------------------------------------------------------
__global__ __launch_bounds__(256) void k_deg_init(float* __restrict__ deg, int n) {
  int i = blockIdx.x * 256 + threadIdx.x;
  if (i < n) deg[i] = 1.0f;               // self-loop weight
}

__global__ __launch_bounds__(256) void k_deg_acc(const long long* __restrict__ dst,
                                                 const float* __restrict__ w,
                                                 float* __restrict__ deg, int e) {
  int i = blockIdx.x * 256 + threadIdx.x;
  if (i < e) atomicAdd(&deg[dst[i]], w[i]);
}

__global__ __launch_bounds__(256) void k_dinv(float* __restrict__ deg, int n) {
  int i = blockIdx.x * 256 + threadIdx.x;
  if (i < n) { float d = deg[i]; deg[i] = (d > 0.0f) ? rsqrtf(d) : 0.0f; }
}

__global__ __launch_bounds__(256) void k_norm(const long long* __restrict__ src,
                                              const long long* __restrict__ dst,
                                              const float* __restrict__ w,
                                              const float* __restrict__ dinv,
                                              float* __restrict__ nrm, int e) {
  int i = blockIdx.x * 256 + threadIdx.x;
  if (i < e) nrm[i] = dinv[src[i]] * w[i] * dinv[dst[i]];
}

// ---------------------------------------------------------------------------
// fp32 -> bf16 converts (vec4)
// ---------------------------------------------------------------------------
__global__ __launch_bounds__(256) void k_cvt_bf16(const float* __restrict__ in,
                                                  __bf16* __restrict__ out, long long n4) {
  long long t = (long long)blockIdx.x * 256 + threadIdx.x;
  if (t >= n4) return;
  float4 v = ((const float4*)in)[t];
  v4bf r = { f2bf(v.x), f2bf(v.y), f2bf(v.z), f2bf(v.w) };
  ((v4bf*)out)[t] = r;
}

__global__ __launch_bounds__(256) void k_relu_cvt_bf16(const float* __restrict__ in,
                                                       __bf16* __restrict__ out, long long n4) {
  long long t = (long long)blockIdx.x * 256 + threadIdx.x;
  if (t >= n4) return;
  float4 v = ((const float4*)in)[t];
  v4bf r = { f2bf(fmaxf(v.x, 0.0f)), f2bf(fmaxf(v.y, 0.0f)),
             f2bf(fmaxf(v.z, 0.0f)), f2bf(fmaxf(v.w, 0.0f)) };
  ((v4bf*)out)[t] = r;
}

__global__ __launch_bounds__(256) void k_relu_inplace(float* __restrict__ x, long long n4) {
  long long t = (long long)blockIdx.x * 256 + threadIdx.x;
  if (t >= n4) return;
  float4 v = ((float4*)x)[t];
  v.x = fmaxf(v.x, 0.0f); v.y = fmaxf(v.y, 0.0f);
  v.z = fmaxf(v.z, 0.0f); v.w = fmaxf(v.w, 0.0f);
  ((float4*)x)[t] = v;
}

// ---------------------------------------------------------------------------
// Pack W [K, N] fp32 row-major into WMMA B fragments (bf16):
// B 32x16 tile, lane l = column l%16, element e -> K = (e/8)*16 + (l/16)*8 + e%8.
// Fragment storage: [tile = tn*KT + tk][lane 0..31][e 0..15] contiguous.
// ---------------------------------------------------------------------------
template <int NT, int KT>
__global__ __launch_bounds__(256) void k_pack_w(const float* __restrict__ W,
                                                __bf16* __restrict__ out) {
  int t = blockIdx.x * 256 + threadIdx.x;
  const int total = NT * KT * 512;        // 32 lanes * 16 elems per tile
  if (t >= total) return;
  int tile = t >> 9;
  int r    = t & 511;
  int lane = r >> 4;
  int e    = r & 15;
  int tn   = tile / KT;
  int tk   = tile % KT;
  int k    = tk * 32 + (e >> 3) * 16 + (lane >> 4) * 8 + (e & 7);
  int n    = tn * 16 + (lane & 15);
  out[t] = f2bf(W[(size_t)k * (NT * 16) + n]);
}

// ---------------------------------------------------------------------------
// WMMA GEMM: C[M, NT*16] = A[M, KT*32] (bf16, row-major) * Bp (packed frags)
// One wave -> 16 rows x full N strip. 8 waves / block. M = Mtiles*16 exact.
// A fragment (16x32 bf16): lane l row l%16; elems 0..7 -> K=(l/16)*8+0..7,
// elems 8..15 -> +16  => two contiguous 16B loads per lane.
// ---------------------------------------------------------------------------
template <int NT, int KT>
__global__ __launch_bounds__(256) void k_wmma_gemm(const __bf16* __restrict__ A,
                                                   const __bf16* __restrict__ Bp,
                                                   float* __restrict__ C, int Mtiles) {
  const int wave  = threadIdx.x >> 5;
  const int lane  = threadIdx.x & 31;
  const int mtile = blockIdx.x * 8 + wave;
  if (mtile >= Mtiles) return;            // whole-wave exit: EXEC stays all-1s

  const int K    = KT * 32;
  const int Nout = NT * 16;
  const int arow = mtile * 16 + (lane & 15);
  const int koff = (lane >> 4) * 8;

  v8f acc[NT] = {};

#pragma unroll
  for (int kt = 0; kt < KT; ++kt) {
    const __bf16* ap = A + (size_t)arow * K + kt * 32 + koff;
    v8bf lo = *(const v8bf*)(ap);
    v8bf hi = *(const v8bf*)(ap + 16);
    v16bf a = __builtin_shufflevector(lo, hi, 0, 1, 2, 3, 4, 5, 6, 7,
                                              8, 9, 10, 11, 12, 13, 14, 15);
#pragma unroll
    for (int nt = 0; nt < NT; ++nt) {
      v16bf b = *(const v16bf*)(Bp + ((size_t)(nt * KT + kt) * 32 + lane) * 16);
      acc[nt] = __builtin_amdgcn_wmma_f32_16x16x32_bf16(
          /*neg_a=*/false, a, /*neg_b=*/false, b,
          /*c_mod=*/(short)0, acc[nt], /*reuse_a=*/false, /*reuse_b=*/false);
    }
  }

  // C/D layout: VGPR v, lanes 0-15 -> row v, lanes 16-31 -> row v+8; col = l%16
  const int crow0 = mtile * 16 + (lane >> 4) * 8;
  const int ccol  = lane & 15;
#pragma unroll
  for (int nt = 0; nt < NT; ++nt) {
#pragma unroll
    for (int v = 0; v < 8; ++v) {
      C[(size_t)(crow0 + v) * Nout + nt * 16 + ccol] = acc[nt][v];
    }
  }
}

// ---------------------------------------------------------------------------
// Aggregation: out[i] = bias + dinv[i]^2 * H[i]   (self-loop fused with init)
// then per-edge: out[dst] += norm[e] * H[src]     (float4, atomicAdd)
// ---------------------------------------------------------------------------
template <int D>
__global__ __launch_bounds__(256) void k_agg_init(const float* __restrict__ H,
                                                  const float* __restrict__ dinv,
                                                  const float* __restrict__ bias,
                                                  float* __restrict__ agg, long long n4) {
  long long t = (long long)blockIdx.x * 256 + threadIdx.x;
  if (t >= n4) return;
  long long idx = t * 4;
  int node = (int)(idx / D);
  int d    = (int)(idx % D);
  float s  = dinv[node]; s *= s;
  float4 h = *(const float4*)(H + idx);
  float4 o;
  o.x = bias[d + 0] + h.x * s;
  o.y = bias[d + 1] + h.y * s;
  o.z = bias[d + 2] + h.z * s;
  o.w = bias[d + 3] + h.w * s;
  *(float4*)(agg + idx) = o;
}

template <int DV4>  // D/4 floats-per-edge groups (power of two -> shifts)
__global__ __launch_bounds__(256) void k_scatter(const float* __restrict__ H,
                                                 const long long* __restrict__ src,
                                                 const long long* __restrict__ dst,
                                                 const float* __restrict__ nrm,
                                                 float* __restrict__ agg, long long total) {
  long long t = (long long)blockIdx.x * 256 + threadIdx.x;
  if (t >= total) return;
  long long e = t / DV4;
  int d4      = (int)(t % DV4);
  long long s = src[e];
  long long d = dst[e];
  float w     = nrm[e];
  const int D = DV4 * 4;
  float4 h = *(const float4*)(H + s * D + d4 * 4);
  float* o = agg + d * D + d4 * 4;
  atomicAdd(o + 0, h.x * w);
  atomicAdd(o + 1, h.y * w);
  atomicAdd(o + 2, h.z * w);
  atomicAdd(o + 3, h.w * w);
}

// ---------------------------------------------------------------------------
// Host launcher
// ---------------------------------------------------------------------------
extern "C" void kernel_launch(void* const* d_in, const int* in_sizes, int n_in,
                              void* d_out, int out_size, void* d_ws, size_t ws_size,
                              hipStream_t stream) {
  (void)in_sizes; (void)n_in; (void)out_size; (void)ws_size;

  const int N = 50000, E = 800000, DIN = 256, D1 = 128, D2 = 64;

  const float*      x   = (const float*)d_in[0];
  const long long*  ei  = (const long long*)d_in[1];   // int64 [2, E]
  const float*      ew  = (const float*)d_in[2];
  const float*      W1  = (const float*)d_in[3];
  const float*      b1  = (const float*)d_in[4];
  const float*      W2  = (const float*)d_in[5];
  const float*      b2  = (const float*)d_in[6];
  const long long*  src = ei;
  const long long*  dst = ei + E;
  float*            out = (float*)d_out;               // [N, 64] f32

  // ---- carve workspace (256B aligned chunks) ----
  char* p = (char*)d_ws;
  auto alloc = [&](size_t bytes) -> void* {
    void* r = (void*)p;
    p += (bytes + 255) & ~(size_t)255;
    return r;
  };
  float*  dinv = (float*)alloc((size_t)N * 4);          // deg -> dinv in place
  float*  nrm  = (float*)alloc((size_t)E * 4);
  __bf16* Xb   = (__bf16*)alloc((size_t)N * DIN * 2);
  __bf16* W1p  = (__bf16*)alloc((size_t)DIN * D1 * 2);
  __bf16* W2p  = (__bf16*)alloc((size_t)D1 * D2 * 2);
  float*  H1   = (float*)alloc((size_t)N * D1 * 4);
  float*  AGG1 = (float*)alloc((size_t)N * D1 * 4);
  __bf16* A2b  = (__bf16*)alloc((size_t)N * D1 * 2);
  float*  H2   = (float*)alloc((size_t)N * D2 * 4);

  auto blocks = [](long long work) { return (unsigned)((work + 255) / 256); };

  // ---- gcn_norm ----
  k_deg_init<<<blocks(N), 256, 0, stream>>>(dinv, N);
  k_deg_acc<<<blocks(E), 256, 0, stream>>>(dst, ew, dinv, E);
  k_dinv<<<blocks(N), 256, 0, stream>>>(dinv, N);
  k_norm<<<blocks(E), 256, 0, stream>>>(src, dst, ew, dinv, nrm, E);

  // ---- convert / pack ----
  k_cvt_bf16<<<blocks((long long)N * DIN / 4), 256, 0, stream>>>(x, Xb, (long long)N * DIN / 4);
  k_pack_w<8, 8><<<blocks(8 * 8 * 512), 256, 0, stream>>>(W1, W1p);   // 256x128
  k_pack_w<4, 4><<<blocks(4 * 4 * 512), 256, 0, stream>>>(W2, W2p);   // 128x64

  const int Mtiles = N / 16;  // 3125, exact
  const unsigned gemm_blocks = (unsigned)((Mtiles + 7) / 8);

  // ---- layer 1: H1 = Xb @ W1 ; AGG1 = scatter + b1 ; relu -> bf16 ----
  k_wmma_gemm<8, 8><<<gemm_blocks, 256, 0, stream>>>(Xb, W1p, H1, Mtiles);
  k_agg_init<D1><<<blocks((long long)N * D1 / 4), 256, 0, stream>>>(H1, dinv, b1, AGG1,
                                                                    (long long)N * D1 / 4);
  k_scatter<D1 / 4><<<blocks((long long)E * (D1 / 4)), 256, 0, stream>>>(H1, src, dst, nrm, AGG1,
                                                                         (long long)E * (D1 / 4));
  k_relu_cvt_bf16<<<blocks((long long)N * D1 / 4), 256, 0, stream>>>(AGG1, A2b,
                                                                     (long long)N * D1 / 4);

  // ---- layer 2: H2 = A2b @ W2 ; d_out = scatter + b2 ; relu in place ----
  k_wmma_gemm<4, 4><<<gemm_blocks, 256, 0, stream>>>(A2b, W2p, H2, Mtiles);
  k_agg_init<D2><<<blocks((long long)N * D2 / 4), 256, 0, stream>>>(H2, dinv, b2, out,
                                                                    (long long)N * D2 / 4);
  k_scatter<D2 / 4><<<blocks((long long)E * (D2 / 4)), 256, 0, stream>>>(H2, src, dst, nrm, out,
                                                                         (long long)E * (D2 / 4));
  k_relu_inplace<<<blocks((long long)N * D2 / 4), 256, 0, stream>>>(out, (long long)N * D2 / 4);
}
